// MoE_layer_85023172591911
// MI455X (gfx1250) — matrix operand order repaired and checked by reading, not compiled, not run on hardware
//
#include <hip/hip_runtime.h>

// ---------------------------------------------------------------------------
// MoE layer (top-2 of 4 experts, conv3x3+BN+ReLU) for MI455X / gfx1250.
// Heavy path: implicit-GEMM conv via V_WMMA_F32_16X16X32_BF16 (wave32 WMMA),
// f32->bf16 conversion while staging tiles into LDS. Sparse dispatch by
// uniform per-block early-exit on gate==0 (EXEC all-ones at every WMMA).
// ---------------------------------------------------------------------------

typedef __bf16 bf16x16 __attribute__((ext_vector_type(16)));
typedef float  f32x8   __attribute__((ext_vector_type(8)));

union FragBF { unsigned int u[8]; bf16x16 v; };

static constexpr int kB = 32, kC = 256, kH = 64, kW = 64, kE = 4, kPROJ = 128;
static constexpr int kHW = kH * kW;      // 4096 spatial positions
static constexpr int kK  = kC * 9;       // 2304 = GEMM K (ci,kh,kw)
static constexpr int TM = 128, TN = 128, KT = 32;
static constexpr float kClampMax = 4.6051701859880914f;   // ln(100)

__device__ __forceinline__ unsigned int pack_bf16(float lo, float hi) {
  // round-to-nearest-even f32 -> bf16, pack pair (lower K index in low half)
  unsigned int ul = __float_as_uint(lo);
  unsigned int uh = __float_as_uint(hi);
  ul = (ul + 0x7FFFu + ((ul >> 16) & 1u)) >> 16;
  uh = (uh + 0x7FFFu + ((uh >> 16) & 1u)) >> 16;
  return (ul & 0xFFFFu) | (uh << 16);
}

// ---------------- spatial mean pool: x[32,256,64,64] -> xf[32,256] ----------
__global__ __launch_bounds__(256) void moe_pool(const float* __restrict__ x,
                                                float* __restrict__ xf) {
  int bc = blockIdx.x;
  const float* px = x + (size_t)bc * kHW;
  float s = 0.f;
  for (int i = threadIdx.x; i < kHW; i += 256) s += px[i];
  __shared__ float red[256];
  red[threadIdx.x] = s;
  __syncthreads();
  for (int st = 128; st > 0; st >>= 1) {
    if ((int)threadIdx.x < st) red[threadIdx.x] += red[threadIdx.x + st];
    __syncthreads();
  }
  if (threadIdx.x == 0) xf[bc] = red[0] * (1.0f / (float)kHW);
}

// ---------------- cosine top-2 gate + aux loss (1 block, 128 threads) -------
__global__ __launch_bounds__(128) void moe_gate(const float* __restrict__ xf,
                                                const float* __restrict__ proj_w,
                                                const float* __restrict__ proj_b,
                                                const float* __restrict__ sim,
                                                const float* __restrict__ temperature,
                                                float* __restrict__ gates,
                                                float* __restrict__ loss_out) {
  __shared__ float p[kPROJ];
  __shared__ float red[kPROJ];
  __shared__ float snorm[kE];
  __shared__ float logits[kE];
  __shared__ float imp[kE];
  __shared__ float loadc[kE];
  int t = threadIdx.x;
  if (t < kE) {
    float s2 = 0.f;
    for (int j = 0; j < kPROJ; j++) { float v = sim[j * kE + t]; s2 += v * v; }
    snorm[t] = fmaxf(sqrtf(s2), 1e-12f);
    imp[t] = 0.f; loadc[t] = 0.f;
  }
  float scale = __expf(fminf(temperature[0], kClampMax));
  __syncthreads();
  for (int b = 0; b < kB; b++) {
    float d = proj_b[t];
    const float* xr = xf + b * kC;
    const float* wr = proj_w + t * kC;
    for (int c = 0; c < kC; c++) d = fmaf(xr[c], wr[c], d);
    red[t] = d * d;
    __syncthreads();
    for (int st = kPROJ / 2; st > 0; st >>= 1) {
      if (t < st) red[t] += red[t + st];
      __syncthreads();
    }
    float nrm = fmaxf(sqrtf(red[0]), 1e-12f);
    p[t] = d / nrm;
    __syncthreads();
    if (t < kE) {
      float l = 0.f;
      for (int j = 0; j < kPROJ; j++) l = fmaf(p[j], sim[j * kE + t], l);
      logits[t] = (l / snorm[t]) * scale;
    }
    __syncthreads();
    if (t == 0) {
      int i1 = 0;
      for (int e = 1; e < kE; e++) if (logits[e] > logits[i1]) i1 = e;   // ties: lower idx
      int i2 = (i1 == 0) ? 1 : 0;
      for (int e = 0; e < kE; e++) {
        if (e == i1) continue;
        if (logits[e] > logits[i2]) i2 = e;
      }
      float m = fmaxf(logits[i1], logits[i2]);
      float e1 = __expf(logits[i1] - m), e2 = __expf(logits[i2] - m);
      float inv = 1.0f / (e1 + e2);
      float g1 = e1 * inv, g2 = e2 * inv;
      for (int e = 0; e < kE; e++) gates[b * kE + e] = 0.f;
      gates[b * kE + i1] = g1;
      gates[b * kE + i2] = g2;
      imp[i1] += g1; imp[i2] += g2;
      loadc[i1] += 1.f; loadc[i2] += 1.f;
    }
    __syncthreads();
  }
  if (t == 0) {
    float tot = 0.f;
    {
      float m = (imp[0] + imp[1] + imp[2] + imp[3]) * 0.25f;
      float var = 0.f;
      for (int e = 0; e < kE; e++) { float dd = imp[e] - m; var += dd * dd; }
      var *= (1.0f / 3.0f);                       // unbiased ddof=1
      tot += var / (m * m + 1e-10f);
    }
    {
      float m = (loadc[0] + loadc[1] + loadc[2] + loadc[3]) * 0.25f;
      float var = 0.f;
      for (int e = 0; e < kE; e++) { float dd = loadc[e] - m; var += dd * dd; }
      var *= (1.0f / 3.0f);
      tot += var / (m * m + 1e-10f);
    }
    loss_out[0] = tot * 0.01f;
  }
}

// ---------------- implicit-GEMM conv3x3 + BN + ReLU + gated combine ---------
// im2col sample: K index kg = ci*9 + kh*3 + kw  (matches conv_w [E,Co,Ci,3,3])
__device__ __forceinline__ float xsample(const float* __restrict__ x,
                                         int b, int kg, int h, int w) {
  int ci = kg / 9;
  int r  = kg - ci * 9;
  int hh = h + r / 3 - 1;
  int ww = w + (r % 3) - 1;
  if ((unsigned)hh >= (unsigned)kH || (unsigned)ww >= (unsigned)kW) return 0.f;
  return x[(((size_t)b * kC + ci) * kH + hh) * kW + ww];
}

__global__ __launch_bounds__(256) void moe_conv(const float* __restrict__ x,
                                                const float* __restrict__ conv_w,
                                                const float* __restrict__ conv_b,
                                                const float* __restrict__ bn_gamma,
                                                const float* __restrict__ bn_beta,
                                                const float* __restrict__ bn_mean,
                                                const float* __restrict__ bn_var,
                                                const float* __restrict__ gates,
                                                float* __restrict__ y) {
  // LDS: bf16 pairs, row stride 17 dwords to dodge bank conflicts
  __shared__ unsigned int Asl[TM * 17];   // A[m][k-pair]   (weights, row-major over m)
  __shared__ unsigned int Bsl[TN * 17];   // B[n][k-pair]   (im2col, column-major)
  __shared__ float aS[TM];                // BN scale per co
  __shared__ float bS[TM];                // (bias - mean)*scale + beta per co

  int tid  = threadIdx.x;
  int wave = tid >> 5, lane = tid & 31;
  int wm   = wave & 3;       // 4 wave rows  -> each owns 2 M-tiles (32 co rows)
  int wn   = wave >> 2;      // 2 wave cols  -> each owns 4 N-tiles (64 cols)
  int l16  = lane & 15, hlf = lane >> 4;

  int b     = blockIdx.z;
  int mBase = blockIdx.y * TM;
  int nBase = blockIdx.x * TN;

  f32x8 zero8 = {};
  f32x8 outAcc[2][4];
  for (int im = 0; im < 2; im++)
    for (int in = 0; in < 4; in++) outAcc[im][in] = zero8;

  for (int e = 0; e < kE; e++) {
    float g = gates[b * kE + e];
    if (g == 0.0f) continue;                 // uniform per block: no divergence

    __syncthreads();                         // protect aS/bS + LDS vs prev expert
    if (tid < TM) {
      int idx = e * kC + mBase + tid;
      float sc = bn_gamma[idx] * rsqrtf(bn_var[idx] + 1e-5f);
      aS[tid] = sc;
      bS[tid] = (conv_b[idx] - bn_mean[idx]) * sc + bn_beta[idx];
    }

    f32x8 acc[2][4];
    for (int im = 0; im < 2; im++)
      for (int in = 0; in < 4; in++) acc[im][in] = zero8;

    for (int k0 = 0; k0 < kK; k0 += KT) {
      __syncthreads();
      // --- stage A slab [TM x 32] : f32 weights -> packed bf16 pairs --------
      {
        int row = tid >> 1;
        int jp  = (tid & 1) * 8;             // pair indices [jp, jp+8)
        const float* src = conv_w + ((size_t)(e * kC + mBase + row)) * kK + k0 + jp * 2;
        unsigned int* dst = &Asl[row * 17 + jp];
#pragma unroll
        for (int j = 0; j < 8; j++) dst[j] = pack_bf16(src[2 * j], src[2 * j + 1]);
      }
      // --- stage B slab [32 x TN] : on-the-fly im2col, bf16, column-major ---
      {
        int j  = tid >> 4;                   // k-pair 0..15
        int n0 = (tid & 15) * 8;             // 8 columns each
        int kg = k0 + 2 * j;
#pragma unroll
        for (int nn = 0; nn < 8; nn++) {
          int pp = nBase + n0 + nn;
          int h = pp >> 6, w = pp & 63;
          Bsl[(n0 + nn) * 17 + j] =
              pack_bf16(xsample(x, b, kg, h, w), xsample(x, b, kg + 1, h, w));
        }
      }
      __syncthreads();
      // --- fragments per ISA layout ----------------------------------------
      // A 16x32 bf16: lanes 0-15 hold K{0-7,16-23}, lanes 16-31 K{8-15,24-31}
      FragBF A[2];
#pragma unroll
      for (int im = 0; im < 2; im++) {
        const unsigned int* ap = &Asl[((wm * 2 + im) * 16 + l16) * 17];
#pragma unroll
        for (int i = 0; i < 4; i++) {
          A[im].u[i]     = ap[hlf * 4 + i];
          A[im].u[4 + i] = ap[8 + hlf * 4 + i];
        }
      }
      // B 32x16 bf16: lane = column; lanes 0-15 K{0-7,16-23}, lanes 16-31 K{8-15,24-31}
#pragma unroll
      for (int in = 0; in < 4; in++) {
        FragBF Bf;
        const unsigned int* bp = &Bsl[((wn * 4 + in) * 16 + l16) * 17];
#pragma unroll
        for (int i = 0; i < 8; i++) Bf.u[i] = bp[(i >> 2) * 8 + hlf * 4 + (i & 3)];
#pragma unroll
        for (int im = 0; im < 2; im++)
          acc[im][in] = __builtin_amdgcn_wmma_f32_16x16x32_bf16(
              false, A[im].v, false, Bf.v, (short)0, acc[im][in], false, false);
      }
    }
    // --- epilogue: bias+BN+ReLU, gate-weighted accumulate --------------------
    // C/D layout: vgpr i holds M = i + 8*laneHalf, N = lane&15
#pragma unroll
    for (int im = 0; im < 2; im++) {
      int co0 = (wm * 2 + im) * 16 + hlf * 8;
#pragma unroll
      for (int in = 0; in < 4; in++) {
#pragma unroll
        for (int i = 0; i < 8; i++) {
          float v = fmaf(acc[im][in][i], aS[co0 + i], bS[co0 + i]);
          outAcc[im][in][i] += g * fmaxf(v, 0.f);
        }
      }
    }
  }

  // --- write y[b, co, p] ----------------------------------------------------
#pragma unroll
  for (int im = 0; im < 2; im++) {
#pragma unroll
    for (int in = 0; in < 4; in++) {
      int co = mBase + (wm * 2 + im) * 16 + hlf * 8;
      int pp = nBase + (wn * 4 + in) * 16 + l16;
      float* dst = y + ((size_t)(b * kC + co)) * kHW + pp;
#pragma unroll
      for (int i = 0; i < 8; i++) dst[(size_t)i * kHW] = outAcc[im][in][i];
    }
  }
}

// ---------------------------------------------------------------------------
extern "C" void kernel_launch(void* const* d_in, const int* in_sizes, int n_in,
                              void* d_out, int out_size, void* d_ws, size_t ws_size,
                              hipStream_t stream) {
  (void)in_sizes; (void)n_in; (void)ws_size;
  const float* x           = (const float*)d_in[0];
  const float* proj_w      = (const float*)d_in[1];
  const float* proj_b      = (const float*)d_in[2];
  const float* sim         = (const float*)d_in[3];
  const float* temperature = (const float*)d_in[4];
  const float* conv_w      = (const float*)d_in[5];
  const float* conv_b      = (const float*)d_in[6];
  const float* bn_gamma    = (const float*)d_in[7];
  const float* bn_beta     = (const float*)d_in[8];
  const float* bn_var_     = (const float*)d_in[10];
  const float* bn_mean     = (const float*)d_in[9];

  float* y    = (float*)d_out;
  float* loss = y + (out_size - 1);      // tuple (y, loss) concatenated flat

  float* xf    = (float*)d_ws;           // [32*256] f32
  float* gates = xf + kB * kC;           // [32*4]   f32   (33 KB total scratch)

  moe_pool<<<kB * kC, 256, 0, stream>>>(x, xf);
  moe_gate<<<1, 128, 0, stream>>>(xf, proj_w, proj_b, sim, temperature, gates, loss);
  moe_conv<<<dim3(kHW / TN, kC / TM, kB), 256, 0, stream>>>(
      x, conv_w, conv_b, bn_gamma, bn_beta, bn_mean, bn_var_, gates, y);
}
